// MultiheadLatentAttention_11493332484830
// MI455X (gfx1250) — matrix operand order repaired and checked by reading, not compiled, not run on hardware
//
#include <hip/hip_runtime.h>
#include <hip/hip_bf16.h>

typedef __bf16 bf16_t;
typedef __attribute__((ext_vector_type(16))) __bf16 bf16x16;
typedef __attribute__((ext_vector_type(8)))  __bf16 bf16x8;
typedef __attribute__((ext_vector_type(8)))  float  f32x8;
typedef __attribute__((ext_vector_type(4)))  float  f32x4;

#define B_    2
#define S_    2048
#define DIM_  2048
#define H_    16
#define NOPE_ 128
#define ROPE_ 64
#define HD_   192
#define QLR_  1536
#define KVLR_ 512
#define BS_   (B_ * S_)

// Build a 16-element bf16 WMMA fragment from two 8-element (16B) chunks.
static __device__ __forceinline__ bf16x16 make_frag(const bf16_t* p0, const bf16_t* p1) {
  bf16x8 lo = *(const bf16x8*)p0;
  bf16x8 hi = *(const bf16x8*)p1;
  bf16x16 r;
#pragma unroll
  for (int i = 0; i < 8; ++i) { r[i] = lo[i]; r[i + 8] = hi[i]; }
  return r;
}

// ---------------------------------------------------------------- elementwise
__global__ void k_f32_to_bf16(const float* __restrict__ in, bf16_t* __restrict__ out, int n4) {
  int i = blockIdx.x * blockDim.x + threadIdx.x;
  if (i < n4) {
    f32x4 v = ((const f32x4*)in)[i];
    bf16_t* o = out + (size_t)i * 4;
    o[0] = (bf16_t)v.x; o[1] = (bf16_t)v.y; o[2] = (bf16_t)v.z; o[3] = (bf16_t)v.w;
  }
}

// W[K,N] f32 (row-major) -> Wt[N,K] bf16 (row-major) via LDS tile
__global__ void k_transpose_bf16(const float* __restrict__ W, bf16_t* __restrict__ Wt,
                                 int K, int N) {
  __shared__ float t[32][33];
  const int kb = blockIdx.x * 32, nb = blockIdx.y * 32;
  const int tx = threadIdx.x & 31, ty = threadIdx.x >> 5;  // 256 thr: ty 0..7
#pragma unroll
  for (int i = 0; i < 32; i += 8)
    t[ty + i][tx] = W[(size_t)(kb + ty + i) * N + nb + tx];
  __syncthreads();
#pragma unroll
  for (int i = 0; i < 32; i += 8)
    Wt[(size_t)(nb + ty + i) * K + kb + tx] = (bf16_t)t[tx][ty + i];
}

// ---------------------------------------------------------------- WMMA GEMM
// C[M,N] f32 = A[M,K] bf16 (row major) x Bt[N,K] bf16 (B transposed, row major)
// block = 128 threads (4 waves stacked along M); each wave computes a 64x64 tile
// (4x4 WMMA accumulators -> 16 WMMAs per 32-wide k-step, 0.5KB traffic/WMMA).
__global__ void __launch_bounds__(128)
k_gemm_bf16(const bf16_t* __restrict__ A, const bf16_t* __restrict__ Bt,
            float* __restrict__ C, int M, int N, int K) {
  const int lane = threadIdx.x & 31;
  const int wave = threadIdx.x >> 5;
  const int l16  = lane & 15;
  const int half = lane >> 4;
  const int row0 = blockIdx.x * 256 + wave * 64;
  const int col0 = blockIdx.y * 64;
  const int aOff = half * 8;    // A frag: lanes 0-15 K=0..7/16..23, lanes 16-31 K=8..15/24..31
  const int bOff = half * 8;    // B frag: lanes 0-15 K=0..7/16..23, lanes 16-31 K=8..15/24..31

  f32x8 acc[4][4] = {};
  const bf16_t* ap[4];
  const bf16_t* bp[4];
#pragma unroll
  for (int i = 0; i < 4; ++i) {
    ap[i] = A  + (size_t)(row0 + i * 16 + l16) * K + aOff;
    bp[i] = Bt + (size_t)(col0 + i * 16 + l16) * K + bOff;
  }

  for (int k0 = 0; k0 < K; k0 += 32) {
    bf16x16 a[4], b[4];
#pragma unroll
    for (int i = 0; i < 4; ++i) {
      a[i] = make_frag(ap[i] + k0, ap[i] + k0 + 16);
      b[i] = make_frag(bp[i] + k0, bp[i] + k0 + 16);
    }
    __builtin_prefetch(ap[0] + k0 + 256, 0, 3);
    __builtin_prefetch(bp[0] + k0 + 256, 0, 3);
#pragma unroll
    for (int i = 0; i < 4; ++i)
#pragma unroll
      for (int j = 0; j < 4; ++j)
        acc[i][j] = __builtin_amdgcn_wmma_f32_16x16x32_bf16(false, a[i], false, b[j],
                                                            (short)0, acc[i][j], false, false);
  }
#pragma unroll
  for (int i = 0; i < 4; ++i)
#pragma unroll
    for (int j = 0; j < 4; ++j) {
      float* cp = C + (size_t)(row0 + i * 16 + half * 8) * N + col0 + j * 16 + l16;
#pragma unroll
      for (int r = 0; r < 8; ++r) cp[(size_t)r * N] = acc[i][j][r];
    }
}

// ---------------------------------------------------------------- rmsnorm
__global__ void k_rmsnorm_bf16(const float* __restrict__ in, const float* __restrict__ w,
                               bf16_t* __restrict__ out, int ncols, int instride) {
  const int row = blockIdx.x;
  const float* r = in + (size_t)row * instride;
  float ss = 0.f;
  for (int c = threadIdx.x; c < ncols; c += blockDim.x) { float v = r[c]; ss += v * v; }
#pragma unroll
  for (int off = 16; off; off >>= 1) ss += __shfl_xor(ss, off, 32);
  __shared__ float red[8];
  if ((threadIdx.x & 31) == 0) red[threadIdx.x >> 5] = ss;
  __syncthreads();
  float tot = 0.f;
#pragma unroll
  for (int i = 0; i < 8; ++i) tot += red[i];
  const float rr = rsqrtf(tot / (float)ncols + 1e-5f);
  bf16_t* o = out + (size_t)row * ncols;
  for (int c = threadIdx.x; c < ncols; c += blockDim.x)
    o[c] = (bf16_t)(r[c] * rr * w[c]);
}

// ---------------------------------------------------------------- RoPE / packing
// xq f32 [BS, H*192] -> q_all bf16 [B,H,S,192], rope on last 64, fold 1/sqrt(192)
__global__ void k_rope_q(const float* __restrict__ xq, const float* __restrict__ cosb,
                         const float* __restrict__ sinb, bf16_t* __restrict__ q_all) {
  const int idx = blockIdx.x * blockDim.x + threadIdx.x;  // BS*H*96
  const int p  = idx % 96;
  const int h  = (idx / 96) % H_;
  const int bs = idx / (96 * H_);
  const int s  = bs % S_, b = bs / S_;
  const float scale = 0.07216878364870323f;  // 1/sqrt(192)
  const float* base = xq + (size_t)bs * (H_ * HD_) + h * HD_;
  float o0, o1;
  if (p < 64) {
    o0 = base[2 * p] * scale; o1 = base[2 * p + 1] * scale;
  } else {
    const int i = p - 64;
    float x0 = base[128 + 2 * i], x1 = base[128 + 2 * i + 1];
    float c = cosb[s * 32 + i], sn = sinb[s * 32 + i];
    o0 = (x0 * c - x1 * sn) * scale;
    o1 = (x0 * sn + x1 * c) * scale;
  }
  bf16_t* q = q_all + ((size_t)((b * H_ + h) * S_ + s)) * HD_ + 2 * p;
  q[0] = (bf16_t)o0; q[1] = (bf16_t)o1;
}

// kvd f32 [BS,576] cols 512..575 -> roped, broadcast to all heads of k_all[...,128..191]
__global__ void k_rope_k(const float* __restrict__ kvd, const float* __restrict__ cosb,
                         const float* __restrict__ sinb, bf16_t* __restrict__ k_all) {
  const int idx = blockIdx.x * blockDim.x + threadIdx.x;  // BS*32
  const int i = idx & 31;
  const int bs = idx >> 5;
  const int s = bs % S_, b = bs / S_;
  const float* base = kvd + (size_t)bs * (KVLR_ + ROPE_) + KVLR_;
  float x0 = base[2 * i], x1 = base[2 * i + 1];
  float c = cosb[s * 32 + i], sn = sinb[s * 32 + i];
  bf16_t o0 = (bf16_t)(x0 * c - x1 * sn);
  bf16_t o1 = (bf16_t)(x0 * sn + x1 * c);
#pragma unroll
  for (int h = 0; h < H_; ++h) {
    bf16_t* k = k_all + ((size_t)((b * H_ + h) * S_ + s)) * HD_ + NOPE_ + 2 * i;
    k[0] = o0; k[1] = o1;
  }
}

// kvu f32 [BS, H*256] -> k_all nope part [b,h,s,0..127] and V transposed vt [b,h,d,s]
__global__ void k_split_kvu(const float* __restrict__ kvu, bf16_t* __restrict__ k_all,
                            bf16_t* __restrict__ vt) {
  const int idx = blockIdx.x * blockDim.x + threadIdx.x;  // BS*H*128
  const int d  = idx & 127;
  const int h  = (idx >> 7) & 15;
  const int bs = idx >> 11;
  const int s = bs % S_, b = bs / S_;
  const float* src = kvu + (size_t)bs * (H_ * 2 * NOPE_) + h * (2 * NOPE_);
  k_all[((size_t)((b * H_ + h) * S_ + s)) * HD_ + d]        = (bf16_t)src[d];
  vt[(((size_t)(b * H_ + h)) * NOPE_ + d) * S_ + s]         = (bf16_t)src[NOPE_ + d];
}

// ---------------------------------------------------------------- flash attention
// One wave per 16-row Q tile; K in causal 32-wide tiles; WMMA for QK^T and PV.
__global__ void __launch_bounds__(128)
k_attn(const bf16_t* __restrict__ q_all, const bf16_t* __restrict__ k_all,
       const bf16_t* __restrict__ vt, bf16_t* __restrict__ attnb) {
  __shared__ bf16_t pst[4][16 * 32];           // per-wave 16x32 P staging (C->A relayout)
  const int lane = threadIdx.x & 31, wave = threadIdx.x >> 5;
  const int l16 = lane & 15, half = lane >> 4;
  const int tile = blockIdx.x * 4 + wave;      // 0..4095
  const int qt = tile & 127;                   // S/16 q-tiles per (b,h)
  const int bh = tile >> 7;                    // 0..31
  const int q0 = qt * 16;
  const bf16_t* qb = q_all + (size_t)bh * S_ * HD_;
  const bf16_t* kb = k_all + (size_t)bh * S_ * HD_;
  const bf16_t* vb = vt    + (size_t)bh * NOPE_ * S_;
  const int aOff = half * 8, bOff = half * 8;

  bf16x16 qf[6];
#pragma unroll
  for (int t = 0; t < 6; ++t) {
    const bf16_t* p = qb + (size_t)(q0 + l16) * HD_ + t * 32 + aOff;
    qf[t] = make_frag(p, p + 16);
  }

  f32x8 o[8] = {};
  float m[8], l[8];
#pragma unroll
  for (int r = 0; r < 8; ++r) { m[r] = -1e30f; l[r] = 0.f; }

  bf16_t* ps = &pst[wave][0];

  for (int k0 = 0; k0 < q0 + 16; k0 += 32) {
    f32x8 sc[2] = {};
#pragma unroll
    for (int j = 0; j < 2; ++j) {
      const bf16_t* kr = kb + (size_t)(k0 + j * 16 + l16) * HD_ + bOff;
#pragma unroll
      for (int t = 0; t < 6; ++t) {
        bf16x16 bfr = make_frag(kr + t * 32, kr + t * 32 + 16);
        sc[j] = __builtin_amdgcn_wmma_f32_16x16x32_bf16(false, qf[t], false, bfr,
                                                        (short)0, sc[j], false, false);
      }
    }
    float mnew[8], corr[8];
#pragma unroll
    for (int r = 0; r < 8; ++r) {
      const int row = q0 + half * 8 + r;
#pragma unroll
      for (int j = 0; j < 2; ++j) {
        const int col = k0 + j * 16 + l16;
        float v = sc[j][r];
        sc[j][r] = (col <= row) ? v : -1e30f;   // causal mask (select, no EXEC change)
      }
      float rmax = fmaxf(sc[0][r], sc[1][r]);
#pragma unroll
      for (int off = 1; off < 16; off <<= 1) rmax = fmaxf(rmax, __shfl_xor(rmax, off, 32));
      mnew[r] = fmaxf(m[r], rmax);
      corr[r] = __expf(m[r] - mnew[r]);
      m[r] = mnew[r];
    }
#pragma unroll
    for (int r = 0; r < 8; ++r) {
      float rs = 0.f;
#pragma unroll
      for (int j = 0; j < 2; ++j) {
        float pf = __expf(sc[j][r] - mnew[r]);
        bf16_t pb = (bf16_t)pf;
        ps[(half * 8 + r) * 32 + j * 16 + l16] = pb;  // row-major 16x32 stage
        rs += (float)pb;                              // l matches bf16-quantized P
      }
#pragma unroll
      for (int off = 1; off < 16; off <<= 1) rs += __shfl_xor(rs, off, 32);
      l[r] = l[r] * corr[r] + rs;
    }
#pragma unroll
    for (int t = 0; t < 8; ++t)
#pragma unroll
      for (int r = 0; r < 8; ++r) o[t][r] *= corr[r];
    // same-wave DS ops are in-order: stores above complete before these loads
    bf16x16 pa = make_frag(ps + l16 * 32 + aOff, ps + l16 * 32 + aOff + 16);
#pragma unroll
    for (int t = 0; t < 8; ++t) {
      const bf16_t* vp = vb + (size_t)(t * 16 + l16) * S_ + k0 + bOff;
      bf16x16 bfr = make_frag(vp, vp + 16);
      o[t] = __builtin_amdgcn_wmma_f32_16x16x32_bf16(false, pa, false, bfr,
                                                     (short)0, o[t], false, false);
    }
  }
  const int b = bh / H_, h = bh % H_;
#pragma unroll
  for (int r = 0; r < 8; ++r) {
    const int s = q0 + half * 8 + r;
    bf16_t* op = attnb + ((size_t)(b * S_ + s)) * (H_ * NOPE_) + h * NOPE_ + l16;
    const float inv = 1.f / l[r];
#pragma unroll
    for (int t = 0; t < 8; ++t) op[t * 16] = (bf16_t)(o[t][r] * inv);
  }
}

// ---------------------------------------------------------------- launch
extern "C" void kernel_launch(void* const* d_in, const int* in_sizes, int n_in,
                              void* d_out, int out_size, void* d_ws, size_t ws_size,
                              hipStream_t stream) {
  const float* x     = (const float*)d_in[0];
  const float* wd_q  = (const float*)d_in[1];
  const float* wu_q  = (const float*)d_in[2];
  const float* q_nw  = (const float*)d_in[3];
  const float* wd_kv = (const float*)d_in[4];
  const float* wu_kv = (const float*)d_in[5];
  const float* kv_nw = (const float*)d_in[6];
  const float* wo    = (const float*)d_in[7];
  const float* cosb  = (const float*)d_in[8];
  const float* sinb  = (const float*)d_in[9];
  float* out = (float*)d_out;

  char* ws = (char*)d_ws;
  size_t off = 0;
  auto alloc = [&](size_t bytes) -> char* {
    char* p = ws + off;
    off += (bytes + 255) & ~(size_t)255;
    return p;
  };
  bf16_t* xbf   = (bf16_t*)alloc((size_t)BS_ * DIM_ * 2);
  bf16_t* wdqT  = (bf16_t*)alloc((size_t)QLR_ * DIM_ * 2);
  bf16_t* wuqT  = (bf16_t*)alloc((size_t)H_ * HD_ * QLR_ * 2);
  bf16_t* wdkvT = (bf16_t*)alloc((size_t)(KVLR_ + ROPE_) * DIM_ * 2);
  bf16_t* wukvT = (bf16_t*)alloc((size_t)H_ * 2 * NOPE_ * KVLR_ * 2);
  bf16_t* woT   = (bf16_t*)alloc((size_t)DIM_ * H_ * NOPE_ * 2);
  float*  cq    = (float*) alloc((size_t)BS_ * QLR_ * 4);
  bf16_t* cqn   = (bf16_t*)alloc((size_t)BS_ * QLR_ * 2);
  float*  kvd   = (float*) alloc((size_t)BS_ * (KVLR_ + ROPE_) * 4);
  bf16_t* ckvn  = (bf16_t*)alloc((size_t)BS_ * KVLR_ * 2);
  float*  xq    = (float*) alloc((size_t)BS_ * H_ * HD_ * 4);
  float*  kvu   = (float*) alloc((size_t)BS_ * H_ * 2 * NOPE_ * 4);
  // aliased regions (lifetimes are disjoint in the launch sequence below)
  bf16_t* q_all = (bf16_t*)cq;                                       // 25.2MB in 25.2MB
  bf16_t* k_all = (bf16_t*)xq;                                       // 25.2MB in 50.3MB
  bf16_t* vtb   = (bf16_t*)((char*)xq + (size_t)B_ * H_ * S_ * HD_ * 2);  // +16.8MB
  bf16_t* attnb = (bf16_t*)kvu;                                      // 16.8MB in 67MB

  k_f32_to_bf16<<<(BS_ * DIM_ / 4 + 255) / 256, 256, 0, stream>>>(x, xbf, BS_ * DIM_ / 4);
  k_transpose_bf16<<<dim3(DIM_ / 32, QLR_ / 32), 256, 0, stream>>>(wd_q, wdqT, DIM_, QLR_);
  k_transpose_bf16<<<dim3(QLR_ / 32, (H_ * HD_) / 32), 256, 0, stream>>>(wu_q, wuqT, QLR_, H_ * HD_);
  k_transpose_bf16<<<dim3(DIM_ / 32, (KVLR_ + ROPE_) / 32), 256, 0, stream>>>(wd_kv, wdkvT, DIM_, KVLR_ + ROPE_);
  k_transpose_bf16<<<dim3(KVLR_ / 32, (H_ * 2 * NOPE_) / 32), 256, 0, stream>>>(wu_kv, wukvT, KVLR_, H_ * 2 * NOPE_);
  k_transpose_bf16<<<dim3((H_ * NOPE_) / 32, DIM_ / 32), 256, 0, stream>>>(wo, woT, H_ * NOPE_, DIM_);

  k_gemm_bf16<<<dim3(BS_ / 256, QLR_ / 64), 128, 0, stream>>>(xbf, wdqT, cq, BS_, QLR_, DIM_);
  k_rmsnorm_bf16<<<BS_, 256, 0, stream>>>(cq, q_nw, cqn, QLR_, QLR_);
  k_gemm_bf16<<<dim3(BS_ / 256, (KVLR_ + ROPE_) / 64), 128, 0, stream>>>(xbf, wdkvT, kvd, BS_, KVLR_ + ROPE_, DIM_);
  k_rmsnorm_bf16<<<BS_, 256, 0, stream>>>(kvd, kv_nw, ckvn, KVLR_, KVLR_ + ROPE_);
  k_gemm_bf16<<<dim3(BS_ / 256, (H_ * HD_) / 64), 128, 0, stream>>>(cqn, wuqT, xq, BS_, H_ * HD_, QLR_);
  k_gemm_bf16<<<dim3(BS_ / 256, (H_ * 2 * NOPE_) / 64), 128, 0, stream>>>(ckvn, wukvT, kvu, BS_, H_ * 2 * NOPE_, KVLR_);

  k_rope_q<<<(BS_ * H_ * 96) / 256, 256, 0, stream>>>(xq, cosb, sinb, q_all);
  k_split_kvu<<<(BS_ * H_ * NOPE_) / 256, 256, 0, stream>>>(kvu, k_all, vtb);
  k_rope_k<<<(BS_ * 32) / 256, 256, 0, stream>>>(kvd, cosb, sinb, k_all);

  k_attn<<<(B_ * H_ * (S_ / 16)) / 4, 128, 0, stream>>>(q_all, k_all, vtb, attnb);

  k_gemm_bf16<<<dim3(BS_ / 256, DIM_ / 64), 128, 0, stream>>>(attnb, woT, out, BS_, DIM_, H_ * NOPE_);

  (void)in_sizes; (void)n_in; (void)out_size; (void)ws_size;
}
